// PointNet_features_7765300871542
// MI455X (gfx1250) — compile-verified, matrix operand order checked by reading, not run on hardware
//
#include <hip/hip_runtime.h>
#include <hip/hip_bf16.h>
#include <math.h>

// ---------------------------------------------------------------------------
// PCT feature extractor for MI455X (gfx1250).
// All heavy matmuls (1x1 convs, energy = K^T K, x_r = V*attn) run through one
// WMMA GEMM kernel (v_wmma_f32_16x16x32_f16, f32 accumulate) whose B operand
// is staged into LDS by the Tensor Data Mover (tensor_load_to_lds, TENSORcnt,
// double-buffered) when the builtin exists; otherwise a cooperative
// global->LDS b128 copy. Softmax / L1 col-norm / BatchNorm stay in f32 VALU.
// Workspace requirement: ~600 MB (explicit region reuse below).
// ---------------------------------------------------------------------------

#define BATCH 16
#define NPTS  2048
#define C1DIM 64
#define DKDIM 1024

typedef _Float16 h16x16 __attribute__((ext_vector_type(16)));
typedef float    f32x8  __attribute__((ext_vector_type(8)));

#if __has_builtin(__builtin_amdgcn_tensor_load_to_lds) && \
    __has_builtin(__builtin_amdgcn_s_wait_tensorcnt)
#define HAS_TDM 1
#else
#define HAS_TDM 0
#endif

__device__ __forceinline__ float gelu_erf(float x) {
  return 0.5f * x * (1.0f + erff(x * 0.70710678118654752f));
}

// K-offset of VGPR v (pair base) for 16-bit A/B fragments, per ISA 7.12.2:
// lanes 0-15 hold K = {0..7, 16..23}, lanes 16-31 hold K = {8..15, 24..31}.
__device__ __forceinline__ int frag_koff(int v, int h) {
  return ((v & 4) << 2) + ((v & 3) << 1) + (h << 3);
}

// A fragment from a row-major [M,lda] matrix: lane%16 -> row, K contiguous.
__device__ __forceinline__ h16x16 load_frag_row(const _Float16* __restrict__ A,
                                                int lda, int mBase, int kBase, int lane) {
  int r = lane & 15, h = lane >> 4;
  const _Float16* row = A + (long)(mBase + r) * lda + kBase;
  h16x16 f;
#pragma unroll
  for (int v = 0; v < 8; ++v) {
    int k = frag_koff(v, h);
    f[2 * v]     = row[k];
    f[2 * v + 1] = row[k + 1];
  }
  return f;
}

// Fragment where lane%16 -> column of a [K,lda] row-major matrix (K strided).
// Used for transposed-A (energy) operands.
__device__ __forceinline__ h16x16 load_frag_col(const _Float16* __restrict__ Bm,
                                                int lda, int kBase, int nBase, int lane) {
  int c = nBase + (lane & 15), h = lane >> 4;
  const _Float16* base = Bm + (long)kBase * lda + c;
  h16x16 f;
#pragma unroll
  for (int v = 0; v < 8; ++v) {
    int k = frag_koff(v, h);
    f[2 * v]     = base[(long)k * lda];
    f[2 * v + 1] = base[(long)(k + 1) * lda];
  }
  return f;
}

// B fragment out of an LDS tile laid out [32 k-rows][256 n-cols] f16.
__device__ __forceinline__ h16x16 lds_frag(const _Float16* tile, int col, int lane) {
  int c = col + (lane & 15), h = lane >> 4;
  h16x16 f;
#pragma unroll
  for (int v = 0; v < 8; ++v) {
    int k = frag_koff(v, h);
    f[2 * v]     = tile[k * 256 + c];
    f[2 * v + 1] = tile[(k + 1) * 256 + c];
  }
  return f;
}

__device__ __forceinline__ f32x8 wmma_f16(h16x16 a, h16x16 b, f32x8 c) {
  return __builtin_amdgcn_wmma_f32_16x16x32_f16(false, a, false, b, (short)0, c,
                                                false, false);
}

#define TILE_HALVES (32 * 256)
#define TILE_BYTES  (TILE_HALVES * 2)

#if HAS_TDM
typedef unsigned int u32x4 __attribute__((ext_vector_type(4)));
typedef int          i32x4 __attribute__((ext_vector_type(4)));
typedef int          i32x8 __attribute__((ext_vector_type(8)));

// Issue a TDM 2D tile load: 32 rows x 256 cols of f16 from a row-major
// [K, Nn] tensor slab starting at gsrc, into LDS byte offset ldsOff.
// D# packing per CDNA5 ISA 8.3/8.4 (group0: count/lds/global/type,
// group1: data_size=2B, tensor_dim0=Nn, tensor_dim1=K, tile 256x32,
// tensor_dim0_stride=Nn). Groups 2/3 unused (2D tensor).
__device__ __forceinline__ void tdm_load_tile(const _Float16* gsrc, unsigned ldsOff,
                                              int Nn, int Krows) {
  unsigned long long ga = (unsigned long long)(uintptr_t)gsrc;
  u32x4 g0;
  g0[0] = 1u;                                                   // count=1
  g0[1] = ldsOff;                                               // lds_addr
  g0[2] = (unsigned)ga;                                         // global_addr lo
  g0[3] = (unsigned)((ga >> 32) & 0x01FFFFFFu) | (2u << 30);    // addr hi | type=2
  i32x8 g1;
  g1[0] = (1 << 16);                                            // data_size = 2B
  g1[1] = (int)((Nn & 0xFFFF) << 16);                           // tensor_dim0 lo16
  g1[2] = (int)(((unsigned)Nn >> 16) & 0xFFFFu) | ((Krows & 0xFFFF) << 16);
  g1[3] = (int)(256u << 16);                                    // tile_dim0 = 256
  g1[4] = 32;                                                   // tile_dim1 = 32
  g1[5] = Nn;                                                   // dim0_stride lo32
  g1[6] = 0;
  g1[7] = 0;
  i32x4 z4 = {0, 0, 0, 0};
#if defined(__clang_major__) && (__clang_major__ >= 23)
  i32x8 z8 = {0, 0, 0, 0, 0, 0, 0, 0};
  __builtin_amdgcn_tensor_load_to_lds(g0, g1, z4, z4, z8, 0);
#else
  __builtin_amdgcn_tensor_load_to_lds(g0, g1, z4, z4, 0);
#endif
}
#endif  // HAS_TDM

// Fallback cooperative tile copy (coalesced b128 -> ds_store_b128).
__device__ __forceinline__ void coop_load_tile(const _Float16* __restrict__ gsrc,
                                               _Float16* dst, int Nn, int tid) {
#pragma unroll
  for (int i = 0; i < 4; ++i) {
    int chunk = tid + i * 256;   // 1024 chunks of 16B
    int row = chunk >> 5;
    int col = (chunk & 31) << 3;
    *(uint4*)(dst + row * 256 + col) = *(const uint4*)(gsrc + (long)row * Nn + col);
  }
}

// ---------------------------------------------------------------------------
// Generic batched WMMA GEMM: out[b] = act(A[b?] * B[b][K,N] + bias)
//   A: f16, row-major [M,lda] (aTrans=0) or column-strided A[m,k]=Ab[k*lda+m]
//      (aTrans=1, used for energy = kq^T kq).
//   B: f16 [K,N] per batch, staged 32x256 into double-buffered LDS via TDM.
// Block = 8 waves; wave tile = 16M x 64N; block tile = 32M x 256N.
// Grid = (N/256, ceil(M/32), B).
// ---------------------------------------------------------------------------
__global__ __launch_bounds__(256) void k_gemm(
    const _Float16* __restrict__ A, long strideA, int lda, int aTrans,
    const _Float16* __restrict__ Bm, long strideB,
    const float* __restrict__ bias,
    float* __restrict__ outF, _Float16* __restrict__ outH, long strideOut,
    int M, int K, int Nn, int act) {
  __shared__ __align__(16) _Float16 shB[2][TILE_HALVES];
  const int tid = threadIdx.x;
  const int lane = tid & 31;
  const int wave = tid >> 5;
  const int m0 = blockIdx.y * 32 + (wave >> 2) * 16;
  const int n0b = blockIdx.x * 256;
  const int nw = (wave & 3) * 64;
  const int b = blockIdx.z;
  const bool active = (m0 < M);  // wave-uniform; EXEC all-ones inside
  const _Float16* Ab = A + (long)b * strideA;
  const _Float16* Bb = Bm + (long)b * strideB + n0b;

#if HAS_TDM
  if (wave == 0) tdm_load_tile(Bb, 0u, Nn, K);  // prologue: slab 0 -> buf 0
#else
  coop_load_tile(Bb, &shB[0][0], Nn, tid);
#endif

  f32x8 c0 = {}, c1 = {}, c2 = {}, c3 = {};
  int buf = 0;
  for (int ks = 0; ks < K; ks += 32, buf ^= 1) {
    __syncthreads();  // everyone done reading shB[buf^1] from previous iter
#if HAS_TDM
    if (wave == 0) {
      if (ks + 32 < K) {
        tdm_load_tile(Bb + (long)(ks + 32) * Nn, (unsigned)((buf ^ 1) * TILE_BYTES),
                      Nn, K);
        __builtin_amdgcn_s_wait_tensorcnt(1);  // in-order: slab[buf] complete
      } else {
        __builtin_amdgcn_s_wait_tensorcnt(0);
      }
    }
#else
    if (ks + 32 < K) coop_load_tile(Bb + (long)(ks + 32) * Nn, &shB[buf ^ 1][0], Nn, tid);
#endif
    __syncthreads();  // shB[buf] visible to all waves
    if (active) {
      h16x16 a = aTrans ? load_frag_col(Ab, lda, ks, m0, lane)
                        : load_frag_row(Ab, lda, m0, ks, lane);
      const _Float16* bt = &shB[buf][0];
      h16x16 b0 = lds_frag(bt, nw + 0,  lane);
      h16x16 b1 = lds_frag(bt, nw + 16, lane);
      h16x16 b2 = lds_frag(bt, nw + 32, lane);
      h16x16 b3 = lds_frag(bt, nw + 48, lane);
      c0 = wmma_f16(a, b0, c0);
      c1 = wmma_f16(a, b1, c1);
      c2 = wmma_f16(a, b2, c2);
      c3 = wmma_f16(a, b3, c3);
    }
  }
  if (!active) return;
  const int h = lane >> 4, col = lane & 15;
#pragma unroll
  for (int v = 0; v < 8; ++v) {
    int row = m0 + v + 8 * h;
    float bv = bias ? bias[row] : 0.0f;
    float x0 = c0[v] + bv, x1 = c1[v] + bv, x2 = c2[v] + bv, x3 = c3[v] + bv;
    if (act) { x0 = gelu_erf(x0); x1 = gelu_erf(x1); x2 = gelu_erf(x2); x3 = gelu_erf(x3); }
    long base = (long)b * strideOut + (long)row * Nn + n0b + nw + col;
    if (outF) {
      outF[base] = x0; outF[base + 16] = x1; outF[base + 32] = x2; outF[base + 48] = x3;
    }
    if (outH) {
      outH[base] = (_Float16)x0; outH[base + 16] = (_Float16)x1;
      outH[base + 32] = (_Float16)x2; outH[base + 48] = (_Float16)x3;
    }
  }
}

// In-place row softmax over P[b][row][0..Nn). One block per row (Nn == 2048).
__global__ __launch_bounds__(256) void k_softmax_rows(_Float16* __restrict__ P, int Nn) {
  _Float16* p = P + ((long)blockIdx.y * Nn + blockIdx.x) * Nn;
  float e[8];
  float mx = -1e30f;
#pragma unroll
  for (int i = 0; i < 8; ++i) {
    e[i] = (float)p[threadIdx.x + i * 256];
    mx = fmaxf(mx, e[i]);
  }
  __shared__ float sh[256];
  sh[threadIdx.x] = mx;
  __syncthreads();
  for (int off = 128; off > 0; off >>= 1) {
    if (threadIdx.x < off) sh[threadIdx.x] = fmaxf(sh[threadIdx.x], sh[threadIdx.x + off]);
    __syncthreads();
  }
  float rmax = sh[0];
  __syncthreads();
  float s = 0.0f;
#pragma unroll
  for (int i = 0; i < 8; ++i) { e[i] = __expf(e[i] - rmax); s += e[i]; }
  sh[threadIdx.x] = s;
  __syncthreads();
  for (int off = 128; off > 0; off >>= 1) {
    if (threadIdx.x < off) sh[threadIdx.x] += sh[threadIdx.x + off];
    __syncthreads();
  }
  float inv = 1.0f / sh[0];
#pragma unroll
  for (int i = 0; i < 8; ++i) p[threadIdx.x + i * 256] = (_Float16)(e[i] * inv);
}

// colsum over rows n of P -> invcol[b][m] = 1/(1e-9 + sum)
__global__ __launch_bounds__(256) void k_colsum(const _Float16* __restrict__ P,
                                                float* __restrict__ invcol, int Nn) {
  int m = blockIdx.x * 256 + threadIdx.x;
  int b = blockIdx.y;
  const _Float16* p = P + (long)b * Nn * Nn + m;
  float s = 0.0f;
  for (int n = 0; n < Nn; ++n) s += (float)p[(long)n * Nn];
  invcol[b * Nn + m] = 1.0f / (1e-9f + s);
}

// y = x - U * invcol  (f16 out, feeds the trans-conv GEMM)
__global__ __launch_bounds__(256) void k_xr_sub(const float* __restrict__ X,
                                                const float* __restrict__ U,
                                                const float* __restrict__ invcol,
                                                _Float16* __restrict__ Yh,
                                                int C, int Nn, int total) {
  int i = blockIdx.x * 256 + threadIdx.x;
  if (i >= total) return;
  int n = i % Nn;
  int b = i / (C * Nn);
  float y = X[i] - U[i] * invcol[b * Nn + n];
  Yh[i] = (_Float16)y;
}

// Batch-stats (biased var) per channel over (B,N)
__global__ __launch_bounds__(256) void k_bn_stats(const float* __restrict__ T,
                                                  float* __restrict__ mean,
                                                  float* __restrict__ var,
                                                  int C, int Nn) {
  int c = blockIdx.x;
  const int total = BATCH * Nn;
  float s = 0.0f, ss = 0.0f;
  for (int i = threadIdx.x; i < total; i += 256) {
    int b = i / Nn, n = i % Nn;
    float x = T[((long)b * C + c) * Nn + n];
    s += x; ss += x * x;
  }
  __shared__ float sh1[256], sh2[256];
  sh1[threadIdx.x] = s; sh2[threadIdx.x] = ss;
  __syncthreads();
  for (int off = 128; off > 0; off >>= 1) {
    if (threadIdx.x < off) {
      sh1[threadIdx.x] += sh1[threadIdx.x + off];
      sh2[threadIdx.x] += sh2[threadIdx.x + off];
    }
    __syncthreads();
  }
  if (threadIdx.x == 0) {
    float m = sh1[0] / total;
    mean[c] = m;
    var[c] = sh2[0] / total - m * m;
  }
}

// out = x + gelu(BN(t)) ; optional f16 copy for the next GEMM
__global__ __launch_bounds__(256) void k_bn_gelu_res(
    const float* __restrict__ X, const float* __restrict__ T,
    const float* __restrict__ mean, const float* __restrict__ var,
    const float* __restrict__ gamma, const float* __restrict__ beta,
    float* __restrict__ outF, _Float16* __restrict__ outH,
    int C, int Nn, int total) {
  int i = blockIdx.x * 256 + threadIdx.x;
  if (i >= total) return;
  int c = (i / Nn) % C;
  float t = T[i];
  float tn = gamma[c] * (t - mean[c]) * rsqrtf(var[c] + 1e-5f) + beta[c];
  float o = X[i] + gelu_erf(tn);
  outF[i] = o;
  if (outH) outH[i] = (_Float16)o;
}

// points [B,N,3] -> gelu(conv 3->64) -> gelu(conv 64->64) -> X1 (f32 + f16)
__global__ __launch_bounds__(256) void k_h1(const float* __restrict__ pts,
                                            const float* __restrict__ w1a,
                                            const float* __restrict__ b1a,
                                            const float* __restrict__ w1b,
                                            const float* __restrict__ b1b,
                                            float* __restrict__ X1,
                                            _Float16* __restrict__ X1h, int Nn) {
  int idx = blockIdx.x * 256 + threadIdx.x;
  int b = idx / Nn, n = idx % Nn;
  const float* p = pts + (long)(b * Nn + n) * 3;
  float p0 = p[0], p1 = p[1], p2 = p[2];
  float hbuf[64];
#pragma unroll
  for (int o = 0; o < 64; ++o)
    hbuf[o] = gelu_erf(w1a[o * 3] * p0 + w1a[o * 3 + 1] * p1 + w1a[o * 3 + 2] * p2 + b1a[o]);
  for (int o = 0; o < 64; ++o) {
    float a = b1b[o];
    const float* wr = w1b + o * 64;
#pragma unroll
    for (int c = 0; c < 64; ++c) a += wr[c] * hbuf[c];
    a = gelu_erf(a);
    long oi = ((long)b * 64 + o) * Nn + n;
    X1[oi] = a;
    X1h[oi] = (_Float16)a;
  }
}

// max over N -> [B,C]
__global__ __launch_bounds__(256) void k_maxpool(const float* __restrict__ X,
                                                 float* __restrict__ out, int Nn) {
  int bc = blockIdx.x;
  const float* row = X + (long)bc * Nn;
  float m = -1e30f;
  for (int i = threadIdx.x; i < Nn; i += 256) m = fmaxf(m, row[i]);
  __shared__ float sh[256];
  sh[threadIdx.x] = m;
  __syncthreads();
  for (int off = 128; off > 0; off >>= 1) {
    if (threadIdx.x < off) sh[threadIdx.x] = fmaxf(sh[threadIdx.x], sh[threadIdx.x + off]);
    __syncthreads();
  }
  if (threadIdx.x == 0) out[bc] = sh[0];
}

__global__ void k_f32_to_f16(const float* __restrict__ s, _Float16* __restrict__ d, int n) {
  int i = blockIdx.x * 256 + threadIdx.x;
  if (i < n) d[i] = (_Float16)s[i];
}

__global__ void k_zero_f16(_Float16* __restrict__ d, int n) {
  int i = blockIdx.x * 256 + threadIdx.x;
  if (i < n) d[i] = (_Float16)0.0f;
}

// ---------------------------------------------------------------------------
extern "C" void kernel_launch(void* const* d_in, const int* in_sizes, int n_in,
                              void* d_out, int out_size, void* d_ws, size_t ws_size,
                              hipStream_t stream) {
  (void)in_sizes; (void)n_in; (void)out_size; (void)ws_size;
  const int Nn = NPTS, Bb = BATCH;

  const float* pts       = (const float*)d_in[0];
  const float* w_h1a     = (const float*)d_in[1];
  const float* b_h1a     = (const float*)d_in[2];
  const float* w_h1b     = (const float*)d_in[3];
  const float* b_h1b     = (const float*)d_in[4];
  const float* a64_wqk   = (const float*)d_in[5];
  const float* a64_wv    = (const float*)d_in[6];
  const float* a64_bv    = (const float*)d_in[7];
  const float* a64_wt    = (const float*)d_in[8];
  const float* a64_bt    = (const float*)d_in[9];
  const float* a64_gamma = (const float*)d_in[10];
  const float* a64_beta  = (const float*)d_in[11];
  const float* w_h2a     = (const float*)d_in[12];
  const float* b_h2a     = (const float*)d_in[13];
  const float* w_h2b     = (const float*)d_in[14];
  const float* b_h2b     = (const float*)d_in[15];
  const float* w_h2c     = (const float*)d_in[16];
  const float* b_h2c     = (const float*)d_in[17];
  const float* a1k_wqk   = (const float*)d_in[18];
  const float* a1k_wv    = (const float*)d_in[19];
  const float* a1k_bv    = (const float*)d_in[20];
  const float* a1k_wt    = (const float*)d_in[21];
  const float* a1k_bt    = (const float*)d_in[22];
  const float* a1k_gamma = (const float*)d_in[23];
  const float* a1k_beta  = (const float*)d_in[24];
  float* out = (float*)d_out;

  // ---- workspace layout (explicit reuse; ~600 MB peak) ----
  char* ws = (char*)d_ws;
  size_t off = 0;
  auto alloc = [&](size_t bytes) -> char* {
    char* p = ws + off;
    off += (bytes + 255) & ~(size_t)255;
    return p;
  };
  _Float16* P   = (_Float16*)alloc((size_t)Bb * Nn * Nn * 2);      // R1: P, later T
  float*    Tt  = (float*)P;                                       //   (same bytes)
  float*    U   = (float*)alloc((size_t)Bb * DKDIM * Nn * 4);      // R2: U, later X4
  float*    X4  = U;
  float*    X3  = (float*)alloc((size_t)Bb * DKDIM * Nn * 4);      // R3
  _Float16* X3h = (_Float16*)alloc((size_t)Bb * DKDIM * Nn * 2);   // R4: X3h, later Yh
  _Float16* Yh  = X3h;
  _Float16* Vh  = (_Float16*)alloc((size_t)Bb * DKDIM * Nn * 2);   // R5
  float*    X1  = (float*)alloc((size_t)Bb * C1DIM * Nn * 4);
  _Float16* X1h = (_Float16*)alloc((size_t)Bb * C1DIM * Nn * 2);
  float*    X2  = (float*)alloc((size_t)Bb * C1DIM * Nn * 4);
  _Float16* X2h = (_Float16*)alloc((size_t)Bb * C1DIM * Nn * 2);
  _Float16* G1  = (_Float16*)alloc((size_t)Bb * C1DIM * Nn * 2);
  _Float16* G2  = (_Float16*)alloc((size_t)Bb * 128 * Nn * 2);
  _Float16* KQh = (_Float16*)alloc((size_t)Bb * 256 * Nn * 2);     // [B,32,N] pad @64 stage
  float* invcol = (float*)alloc((size_t)Bb * Nn * 4);
  float* bnmean = (float*)alloc(4096);
  float* bnvar  = (float*)alloc(4096);
  _Float16* wqk64h = (_Float16*)alloc(16 * 64 * 2);
  _Float16* wv64h  = (_Float16*)alloc(64 * 64 * 2);
  _Float16* wt64h  = (_Float16*)alloc(64 * 64 * 2);
  _Float16* wh2ah  = (_Float16*)alloc(64 * 64 * 2);
  _Float16* wh2bh  = (_Float16*)alloc(128 * 64 * 2);
  _Float16* wh2ch  = (_Float16*)alloc(1024 * 128 * 2);
  _Float16* wqk1kh = (_Float16*)alloc(256 * 1024 * 2);
  _Float16* wv1kh  = (_Float16*)alloc((size_t)1024 * 1024 * 2);
  _Float16* wt1kh  = (_Float16*)alloc((size_t)1024 * 1024 * 2);

  auto cvt = [&](const float* s, _Float16* d, int n) {
    k_f32_to_f16<<<dim3((n + 255) / 256), dim3(256), 0, stream>>>(s, d, n);
  };
  auto gemm = [&](const _Float16* A, long sA, int lda, int aT,
                  const _Float16* Bm, long sB, const float* bias,
                  float* oF, _Float16* oH, long sO, int M, int K, int act) {
    dim3 grid(Nn / 256, (M + 31) / 32, Bb);
    k_gemm<<<grid, dim3(256), 0, stream>>>(A, sA, lda, aT, Bm, sB, bias,
                                           oF, oH, sO, M, K, Nn, act);
  };
  auto elts = [&](int total) { return dim3((total + 255) / 256); };

  // ---- f16 weight conversion ----
  cvt(a64_wqk, wqk64h, 16 * 64);
  cvt(a64_wv,  wv64h,  64 * 64);
  cvt(a64_wt,  wt64h,  64 * 64);
  cvt(w_h2a,   wh2ah,  64 * 64);
  cvt(w_h2b,   wh2bh,  128 * 64);
  cvt(w_h2c,   wh2ch,  1024 * 128);
  cvt(a1k_wqk, wqk1kh, 256 * 1024);
  cvt(a1k_wv,  wv1kh,  1024 * 1024);
  cvt(a1k_wt,  wt1kh,  1024 * 1024);

  // ---- h1: points -> X1 [B,64,N] ----
  k_h1<<<dim3((Bb * Nn) / 256), dim3(256), 0, stream>>>(pts, w_h1a, b_h1a, w_h1b, b_h1b,
                                                        X1, X1h, Nn);

  // ---- offset attention @64 (kq padded to 32 rows so K%32==0) ----
  k_zero_f16<<<elts(Bb * 32 * Nn), dim3(256), 0, stream>>>(KQh, Bb * 32 * Nn);
  gemm(wqk64h, 0, 64, 0, X1h, (long)64 * Nn, nullptr, nullptr, KQh, (long)32 * Nn, 16, 64, 0);
  gemm(wv64h, 0, 64, 0, X1h, (long)64 * Nn, a64_bv, nullptr, Vh, (long)64 * Nn, 64, 64, 0);
  gemm(KQh, (long)32 * Nn, Nn, 1, KQh, (long)32 * Nn, nullptr, nullptr, P,
       (long)Nn * Nn, Nn, 32, 0);  // energy = kq^T kq
  k_softmax_rows<<<dim3(Nn, Bb), dim3(256), 0, stream>>>(P, Nn);
  k_colsum<<<dim3(Nn / 256, Bb), dim3(256), 0, stream>>>(P, invcol, Nn);
  gemm(Vh, (long)64 * Nn, Nn, 0, P, (long)Nn * Nn, nullptr, U, nullptr,
       (long)64 * Nn, 64, Nn, 0);
  k_xr_sub<<<elts(Bb * 64 * Nn), dim3(256), 0, stream>>>(X1, U, invcol, Yh, 64, Nn, Bb * 64 * Nn);
  gemm(wt64h, 0, 64, 0, Yh, (long)64 * Nn, a64_bt, Tt, nullptr, (long)64 * Nn, 64, 64, 0);
  k_bn_stats<<<dim3(64), dim3(256), 0, stream>>>(Tt, bnmean, bnvar, 64, Nn);
  k_bn_gelu_res<<<elts(Bb * 64 * Nn), dim3(256), 0, stream>>>(
      X1, Tt, bnmean, bnvar, a64_gamma, a64_beta, X2, X2h, 64, Nn, Bb * 64 * Nn);

  // ---- h2: 64 -> 64 -> 128 -> 1024 (GELU each) ----
  gemm(wh2ah, 0, 64, 0, X2h, (long)64 * Nn, b_h2a, nullptr, G1, (long)64 * Nn, 64, 64, 1);
  gemm(wh2bh, 0, 64, 0, G1, (long)64 * Nn, b_h2b, nullptr, G2, (long)128 * Nn, 128, 64, 1);
  gemm(wh2ch, 0, 128, 0, G2, (long)128 * Nn, b_h2c, X3, X3h, (long)1024 * Nn, 1024, 128, 1);

  // ---- offset attention @1024 ----
  gemm(wqk1kh, 0, 1024, 0, X3h, (long)1024 * Nn, nullptr, nullptr, KQh,
       (long)256 * Nn, 256, 1024, 0);
  gemm(wv1kh, 0, 1024, 0, X3h, (long)1024 * Nn, a1k_bv, nullptr, Vh,
       (long)1024 * Nn, 1024, 1024, 0);
  gemm(KQh, (long)256 * Nn, Nn, 1, KQh, (long)256 * Nn, nullptr, nullptr, P,
       (long)Nn * Nn, Nn, 256, 0);  // energy = kq^T kq
  k_softmax_rows<<<dim3(Nn, Bb), dim3(256), 0, stream>>>(P, Nn);
  k_colsum<<<dim3(Nn / 256, Bb), dim3(256), 0, stream>>>(P, invcol, Nn);
  gemm(Vh, (long)1024 * Nn, Nn, 0, P, (long)Nn * Nn, nullptr, U, nullptr,
       (long)1024 * Nn, 1024, Nn, 0);
  k_xr_sub<<<elts(Bb * 1024 * Nn), dim3(256), 0, stream>>>(X3, U, invcol, Yh, 1024, Nn,
                                                           Bb * 1024 * Nn);
  gemm(wt1kh, 0, 1024, 0, Yh, (long)1024 * Nn, a1k_bt, Tt, nullptr,
       (long)1024 * Nn, 1024, 1024, 0);
  k_bn_stats<<<dim3(1024), dim3(256), 0, stream>>>(Tt, bnmean, bnvar, 1024, Nn);
  k_bn_gelu_res<<<elts(Bb * 1024 * Nn), dim3(256), 0, stream>>>(
      X3, Tt, bnmean, bnvar, a1k_gamma, a1k_beta, X4, nullptr, 1024, Nn, Bb * 1024 * Nn);

  // ---- global max pool -> [B,1024] ----
  k_maxpool<<<dim3(Bb * DKDIM), dim3(256), 0, stream>>>(X4, out, Nn);
  (void)G2; (void)G1;
}
